// UniCrystalFormer_74268574482996
// MI455X (gfx1250) — matrix operand order repaired and checked by reading, not compiled, not run on hardware
//
#include <hip/hip_runtime.h>
#include <hip/hip_bf16.h>
#include <math.h>

// ---------------------------------------------------------------------------
// UniCrystalFormer forward for MI455X (gfx1250, wave32, WMMA).
// GEMMs: bf16 operands, f32 accumulate via v_wmma_f32_16x16x32_bf16.
// A tile (incl. gather-concat rows) staged to LDS via
// global_load_async_to_lds_b128 (+ s_wait_asynccnt), shared by all 4 waves.
// Each wave computes a 16x32 C tile (2 accumulators, shared A fragment).
// ---------------------------------------------------------------------------

typedef __bf16 bf16;
typedef __bf16 v16bf __attribute__((ext_vector_type(16)));
typedef __bf16 v8bf  __attribute__((ext_vector_type(8)));
typedef float  v8f   __attribute__((ext_vector_type(8)));

__device__ __forceinline__ bf16 f2bf(float f) {
  union { float f; unsigned u; } a; a.f = f;
  unsigned r = a.u + 0x7FFFu + ((a.u >> 16) & 1u);
  union { unsigned short s; bf16 b; } c; c.s = (unsigned short)(r >> 16);
  return c.b;
}
__device__ __forceinline__ float sigmoidf_(float x) { return 1.f / (1.f + __expf(-x)); }
__device__ __forceinline__ float siluf_(float x) { return x * sigmoidf_(x); }
__device__ __forceinline__ float softplusf_(float x) { return (x > 20.f) ? x : log1pf(__expf(x)); }
__device__ __forceinline__ float act_apply(float x, int act) {
  switch (act) { case 1: return siluf_(x); case 2: return sigmoidf_(x); case 3: return softplusf_(x); default: return x; }
}
__device__ __forceinline__ v8bf ld8(const bf16* p) { return *reinterpret_cast<const v8bf*>(p); }

#define SHUF16(a,b) __builtin_shufflevector(a,b,0,1,2,3,4,5,6,7,8,9,10,11,12,13,14,15)

// async 16B global->LDS copy (per-lane), tracked with ASYNCcnt
__device__ __forceinline__ void async_cp16(unsigned lds_byte, const bf16* g) {
  asm volatile("global_load_async_to_lds_b128 %0, %1, off" :: "v"(lds_byte), "v"(g) : "memory");
}
__device__ __forceinline__ void wait_async0() {
  asm volatile("s_wait_asynccnt 0" ::: "memory");
}

#define A_PAD 8                      // +8 bf16/row: half-wave lanes hit disjoint bank quads
#define A_STRIDE_MAX (384 + A_PAD)   // max K = 384

// ---------------- generic WMMA GEMM: C[M,N] = act(A[M,K] @ W[K,N] + bias) ---
// WT is W transposed to [N,K] bf16. Block = 4 waves -> 16x128 C tile.
__global__ __launch_bounds__(128) void k_gemm(
    const bf16* __restrict__ A, int lda,
    const bf16* __restrict__ WT,
    const float* __restrict__ bias,
    float* __restrict__ C, bf16* __restrict__ Cbf, int ldc,
    int N, int K, int act)
{
  __shared__ __align__(16) bf16 As[16 * A_STRIDE_MAX];
  const int row0 = blockIdx.x << 4;
  const int stride = K + A_PAD;
  const unsigned lbase = (unsigned)(unsigned long long)(const void*)As;

  // cooperative async staging of the 16xK A tile into LDS
  {
    const int cpr = K >> 3;          // 16B chunks per row
    const int total = cpr << 4;      // 16 rows
    for (int c = threadIdx.x; c < total; c += 128) {
      int r = c / cpr, q = c - r * cpr;
      const bf16* g = A + (long long)(row0 + r) * lda + (q << 3);
      unsigned ld = lbase + (unsigned)((r * stride + (q << 3)) << 1);
      async_cp16(ld, g);
    }
  }
  wait_async0();
  __syncthreads();

  const int lane = threadIdx.x & 31, wave = threadIdx.x >> 5;
  const int col0 = (blockIdx.y << 7) + (wave << 5);   // 32 cols per wave
  if (col0 >= N) return;
  const int mr = lane & 15, hi = lane >> 4;
  const int kh = hi << 4, bk = hi << 4;
  const bf16* arow = As + mr * stride;
  const bf16* brow0 = WT + (long long)(col0 + mr) * K;
  const bf16* brow1 = WT + (long long)(col0 + 16 + mr) * K;
  v8f acc0 = {}, acc1 = {};
  for (int kb = 0; kb < K; kb += 32) {
    v8bf a0 = ld8(arow + kb + kh);
    v8bf a1 = ld8(arow + kb + kh + 8);
    v16bf av = SHUF16(a0, a1);
    v8bf b0 = ld8(brow0 + kb + bk);
    v8bf b1 = ld8(brow0 + kb + bk + 8);
    v16bf bv0 = SHUF16(b0, b1);
    v8bf b2 = ld8(brow1 + kb + bk);
    v8bf b3 = ld8(brow1 + kb + bk + 8);
    v16bf bv1 = SHUF16(b2, b3);
    acc0 = __builtin_amdgcn_wmma_f32_16x16x32_bf16(false, av, false, bv0, (short)0, acc0, false, false);
    acc1 = __builtin_amdgcn_wmma_f32_16x16x32_bf16(false, av, false, bv1, (short)0, acc1, false, false);
  }
  const int rb = row0 + (hi << 3);
  const int cc0 = col0 + mr, cc1 = col0 + 16 + mr;
  const float bb0 = bias ? bias[cc0] : 0.f;
  const float bb1 = bias ? bias[cc1] : 0.f;
#pragma unroll
  for (int v = 0; v < 8; ++v) {
    float y0 = act_apply(acc0[v] + bb0, act);
    float y1 = act_apply(acc1[v] + bb1, act);
    long long o = (long long)(rb + v) * ldc;
    if (C)   { C[o + cc0] = y0; C[o + cc1] = y1; }
    if (Cbf) { Cbf[o + cc0] = f2bf(y0); Cbf[o + cc1] = f2bf(y1); }
  }
}

// ------- gather-concat GEMM: A row m = [X0[i0[e]]|X1[i1[e]]|X2[e]] (K=384) --
// m -> e = m>>hshift, head hd = m & ((1<<hshift)-1), column offset hd*128.
__global__ __launch_bounds__(128) void k_gemm_cat3(
    const bf16* __restrict__ X0, int ld0, const bf16* __restrict__ X1, int ld1,
    const bf16* __restrict__ X2, int ld2,
    const int* __restrict__ i0, const int* __restrict__ i1, int hshift,
    const bf16* __restrict__ WT, const float* __restrict__ bias,
    float* __restrict__ C, bf16* __restrict__ Cbf, int ldc, int N, int act)
{
  __shared__ __align__(16) bf16 As[16 * A_STRIDE_MAX];
  const int row0 = blockIdx.x << 4;
  const int stride = 384 + A_PAD;
  const unsigned lbase = (unsigned)(unsigned long long)(const void*)As;

  // cooperative async gather of 16 concat rows (16 x 384) into LDS
  for (int c = threadIdx.x; c < 768; c += 128) {   // 48 chunks x 16 rows
    int r = c / 48, koff = (c - r * 48) << 3;
    int m = row0 + r;
    int eid = m >> hshift;
    int co = (m & ((1 << hshift) - 1)) << 7;
    int seg = koff >> 7, within = koff & 127;
    const bf16* base;
    if (seg == 0)      base = X0 + (long long)i0[eid] * ld0 + co;
    else if (seg == 1) base = X1 + (long long)i1[eid] * ld1 + co;
    else               base = X2 + (long long)eid * ld2 + co;
    unsigned ld = lbase + (unsigned)((r * stride + koff) << 1);
    async_cp16(ld, base + within);
  }
  wait_async0();
  __syncthreads();

  const int lane = threadIdx.x & 31, wave = threadIdx.x >> 5;
  const int col0 = (blockIdx.y << 7) + (wave << 5);
  if (col0 >= N) return;
  const int mr = lane & 15, hi = lane >> 4;
  const int kh = hi << 4, bk = hi << 4;
  const bf16* arow = As + mr * stride;
  const bf16* brow0 = WT + (long long)(col0 + mr) * 384;
  const bf16* brow1 = WT + (long long)(col0 + 16 + mr) * 384;
  v8f acc0 = {}, acc1 = {};
#pragma unroll
  for (int kb = 0; kb < 384; kb += 32) {
    v8bf a0 = ld8(arow + kb + kh);
    v8bf a1 = ld8(arow + kb + kh + 8);
    v16bf av = SHUF16(a0, a1);
    v8bf b0 = ld8(brow0 + kb + bk);
    v8bf b1 = ld8(brow0 + kb + bk + 8);
    v16bf bv0 = SHUF16(b0, b1);
    v8bf b2 = ld8(brow1 + kb + bk);
    v8bf b3 = ld8(brow1 + kb + bk + 8);
    v16bf bv1 = SHUF16(b2, b3);
    acc0 = __builtin_amdgcn_wmma_f32_16x16x32_bf16(false, av, false, bv0, (short)0, acc0, false, false);
    acc1 = __builtin_amdgcn_wmma_f32_16x16x32_bf16(false, av, false, bv1, (short)0, acc1, false, false);
  }
  const int rb = row0 + (hi << 3);
  const int cc0 = col0 + mr, cc1 = col0 + 16 + mr;
  const float bb0 = bias ? bias[cc0] : 0.f;
  const float bb1 = bias ? bias[cc1] : 0.f;
#pragma unroll
  for (int v = 0; v < 8; ++v) {
    float y0 = act_apply(acc0[v] + bb0, act);
    float y1 = act_apply(acc1[v] + bb1, act);
    long long o = (long long)(rb + v) * ldc;
    if (C)   { C[o + cc0] = y0; C[o + cc1] = y1; }
    if (Cbf) { Cbf[o + cc0] = f2bf(y0); Cbf[o + cc1] = f2bf(y1); }
  }
}

// ---------------- small helper kernels -------------------------------------
__global__ void k_transpose(const float* __restrict__ W, bf16* __restrict__ WT,
                            int K, int N)
{
  long long i = (long long)blockIdx.x * blockDim.x + threadIdx.x;
  if (i >= (long long)K * N) return;
  int n = (int)(i / K), k = (int)(i % K);
  WT[i] = f2bf(W[(long long)k * N + n]);
}

__global__ void k_zero(float* p, long long n) {
  long long i = (long long)blockIdx.x * blockDim.x + threadIdx.x;
  if (i < n) p[i] = 0.f;
}

__global__ void k_embed(const int* __restrict__ xt, const float* __restrict__ emb,
                        float* xc, float* xm, bf16* xcb, bf16* xmb, int N)
{
  int i = blockIdx.x * blockDim.x + threadIdx.x;
  if (i >= N * 128) return;
  int n = i >> 7, c = i & 127;
  float v = emb[xt[n] * 128 + c];
  xc[i] = v; xm[i] = v; xcb[i] = f2bf(v); xmb[i] = f2bf(v);
}

__global__ void k_rbf(const float* __restrict__ ea, bf16* __restrict__ rbf, int E)
{
  long long i = (long long)blockIdx.x * blockDim.x + threadIdx.x;
  if (i >= (long long)E * 128) return;
  int e = (int)(i >> 7), b = (int)(i & 127);
  float x0 = ea[e * 3], x1 = ea[e * 3 + 1], x2 = ea[e * 3 + 2];
  float d = sqrtf(x0 * x0 + x1 * x1 + x2 * x2);
  const float step = 8.0f / 127.0f;
  float t = d - (float)b * step;
  float g = 1.0f / (step * step);
  rbf[i] = f2bf(__expf(-g * t * t));
}

// column-wise batch stats (mean, rstd) over M rows of a [M,128] matrix
__global__ __launch_bounds__(256) void k_colstats(const float* __restrict__ X,
                                                  int M, float* mean, float* rstd)
{
  int c = blockIdx.x, t = threadIdx.x;
  float s = 0.f, s2 = 0.f;
  for (int r = t; r < M; r += 256) {
    float v = X[(long long)r * 128 + c]; s += v; s2 += v * v;
  }
  __shared__ float sh1[256], sh2[256];
  sh1[t] = s; sh2[t] = s2; __syncthreads();
  for (int k = 128; k > 0; k >>= 1) {
    if (t < k) { sh1[t] += sh1[t + k]; sh2[t] += sh2[t + k]; }
    __syncthreads();
  }
  if (t == 0) {
    float m = sh1[0] / (float)M;
    float var = sh2[0] / (float)M - m * m;
    mean[c] = m; rstd[c] = rsqrtf(var + 1e-5f);
  }
}

// CartNet edge update: gate_pre=BN(gate); e += gate_pre (f32+bf16 mirror);
// scatter sigmoid(gate_pre)*msgc into agg[dst]
__global__ void k_cart_edge(const float* __restrict__ gate, const float* __restrict__ msgc,
                            const int* __restrict__ dst,
                            const float* __restrict__ mean, const float* __restrict__ rstd,
                            float* e_f, bf16* e_bf, float* agg, int E)
{
  long long i = (long long)blockIdx.x * blockDim.x + threadIdx.x;
  if (i >= (long long)E * 128) return;
  int c = (int)(i & 127), eid = (int)(i >> 7);
  float g = (gate[i] - mean[c]) * rstd[c];
  float en = e_f[i] + g;
  e_f[i] = en; e_bf[i] = f2bf(en);
  atomicAdd(&agg[(long long)dst[eid] * 128 + c], sigmoidf_(g) * msgc[i]);
}

__global__ void k_bn_silu(const float* __restrict__ xc, const float* __restrict__ agg,
                          const float* __restrict__ mean, const float* __restrict__ rstd,
                          float* xcart, int N)
{
  int i = blockIdx.x * blockDim.x + threadIdx.x;
  if (i >= N * 128) return;
  int c = i & 127;
  xcart[i] = xc[i] + siluf_((agg[i] - mean[c]) * rstd[c]);
}

// alpha = q[dst,hd]*key/sqrt(H); msg *= sigmoid(LN(alpha)); write bf16 mirror
__global__ __launch_bounds__(128) void k_alpha(const float* __restrict__ q,
    float* __restrict__ msg, bf16* __restrict__ msg_bf,
    const float* __restrict__ key, const int* __restrict__ dst,
    const float* __restrict__ g, const float* __restrict__ b)
{
  int m = blockIdx.x, c = threadIdx.x;
  int e = m >> 2, hd = m & 3;
  long long ro = (long long)m * 128;
  float a = q[(long long)dst[e] * 512 + hd * 128 + c] * key[ro + c] * 0.08838834764831845f;
  __shared__ float sh[128];
  sh[c] = a; __syncthreads();
  for (int k = 64; k > 0; k >>= 1) { if (c < k) sh[c] += sh[c + k]; __syncthreads(); }
  float mu = sh[0] / 128.f; __syncthreads();
  float d = a - mu;
  sh[c] = d * d; __syncthreads();
  for (int k = 64; k > 0; k >>= 1) { if (c < k) sh[c] += sh[c + k]; __syncthreads(); }
  float var = sh[0] / 128.f;
  float nl = d * rsqrtf(var + 1e-5f) * g[c] + b[c];
  float v = msg[ro + c] * sigmoidf_(nl);
  msg[ro + c] = v; msg_bf[ro + c] = f2bf(v);
}

// y = LN(row)*g+b ; scatter 0.25*y into out[dst[e]] (head-mean folded in)
__global__ __launch_bounds__(128) void k_mlout(const float* __restrict__ t,
    const int* __restrict__ dst, const float* __restrict__ g, const float* __restrict__ b,
    float* __restrict__ outn)
{
  int m = blockIdx.x, c = threadIdx.x;
  int e = m >> 2;
  long long ro = (long long)m * 128;
  float a = t[ro + c];
  __shared__ float sh[128];
  sh[c] = a; __syncthreads();
  for (int k = 64; k > 0; k >>= 1) { if (c < k) sh[c] += sh[c + k]; __syncthreads(); }
  float mu = sh[0] / 128.f; __syncthreads();
  float d = a - mu;
  sh[c] = d * d; __syncthreads();
  for (int k = 64; k > 0; k >>= 1) { if (c < k) sh[c] += sh[c + k]; __syncthreads(); }
  float var = sh[0] / 128.f;
  float y = d * rsqrtf(var + 1e-5f) * g[c] + b[c];
  atomicAdd(&outn[(long long)dst[e] * 128 + c], 0.25f * y);
}

// beta-gated skip mix
__global__ __launch_bounds__(128) void k_beta(const float* __restrict__ outn,
    const float* __restrict__ xr, const float* __restrict__ bw,
    const float* __restrict__ bb, float* xmat)
{
  int n = blockIdx.x, c = threadIdx.x;
  long long o = (long long)n * 128 + c;
  float ov = outn[o], rv = xr[o];
  float p = ov * bw[c] + rv * bw[128 + c] + (ov - rv) * bw[256 + c];
  __shared__ float sh[128];
  sh[c] = p; __syncthreads();
  for (int k = 64; k > 0; k >>= 1) { if (c < k) sh[c] += sh[c + k]; __syncthreads(); }
  float beta = sigmoidf_(sh[0] + bb[0]);
  xmat[o] = beta * rv + (1.f - beta) * ov;
}

__global__ void k_concat2(const float* __restrict__ xcart, const float* __restrict__ xmat,
                          bf16* __restrict__ xx, int N)
{
  int i = blockIdx.x * blockDim.x + threadIdx.x;
  if (i >= N * 128) return;
  int n = i >> 7, c = i & 127;
  xx[(long long)n * 256 + c] = f2bf(xcart[i]);
  xx[(long long)n * 256 + 128 + c] = f2bf(xmat[i]);
}

__global__ void k_mix(const float* __restrict__ gate2, const float* __restrict__ fused,
                      const float* __restrict__ xcart, const float* __restrict__ xmat,
                      float* xout, float* xc, float* xm, bf16* xcb, bf16* xmb, int N)
{
  int i = blockIdx.x * blockDim.x + threadIdx.x;
  if (i >= N * 128) return;
  float g = sigmoidf_(gate2[i]);
  float xo = g * fused[i] + (1.f - g) * 0.5f * (xcart[i] + xmat[i]);
  xout[i] = xo;
  float a = xo + xcart[i], b = xo + xmat[i];
  xc[i] = a; xcb[i] = f2bf(a);
  xm[i] = b; xmb[i] = f2bf(b);
}

__global__ void k_pool(const float* __restrict__ xout, const int* __restrict__ batch,
                       float* sums, float* cnt, int N)
{
  int i = blockIdx.x * blockDim.x + threadIdx.x;
  if (i >= N * 128) return;
  int n = i >> 7, c = i & 127;
  atomicAdd(&sums[(long long)batch[n] * 128 + c], xout[i]);
  if (c == 0) atomicAdd(&cnt[batch[n]], 1.0f);
}

__global__ __launch_bounds__(128) void k_fc(const float* __restrict__ sums,
    const float* __restrict__ cnt, const float* __restrict__ w1,
    const float* __restrict__ b1, const float* __restrict__ w2,
    const float* __restrict__ b2, float* __restrict__ outv)
{
  int g = blockIdx.x, c = threadIdx.x;
  __shared__ float pl[128], sh[128];
  float cv = cnt[g]; if (cv < 1.f) cv = 1.f;
  pl[c] = sums[(long long)g * 128 + c] / cv; __syncthreads();
  float acc = b1[c];
  for (int k = 0; k < 128; ++k) acc += pl[k] * w1[k * 128 + c];
  sh[c] = siluf_(acc) * w2[c]; __syncthreads();
  for (int k = 64; k > 0; k >>= 1) { if (c < k) sh[c] += sh[c + k]; __syncthreads(); }
  if (c == 0) outv[g] = sh[0] + b2[0];
}

// ---------------------------------------------------------------------------
extern "C" void kernel_launch(void* const* d_in, const int* in_sizes, int n_in,
                              void* d_out, int out_size, void* d_ws, size_t ws_size,
                              hipStream_t stream)
{
  (void)n_in; (void)out_size; (void)ws_size;
  const int N = in_sizes[0];        // 4096 nodes
  const int E = in_sizes[1] / 2;    // 32768 edges
  const int H = 128, HC = 512, L = 5, NG = 8;

  const int* xtype = (const int*)d_in[0];
  const int* srcI  = (const int*)d_in[1];
  const int* dstI  = srcI + E;
  const float* eattr = (const float*)d_in[2];
  const int* batch = (const int*)d_in[3];
  auto F = [&](int i) { return (const float*)d_in[i]; };
  // param flatten order (jax pytree, alphabetical nested dict keys):
  // 4 embed, 5 fc_b1, 6 fc_b2, 7 fc_w1, 8 fc_w2,
  // 9 a_ln_b,10 a_ln_g,11 beta_b,12 beta_w,13 ca_b1,14 ca_b2,15 ca_w1,16 ca_w2,
  // 17 cg_b1,18 cg_b2,19 cg_w1,20 cg_w2,21 cm_gate_b,22 cm_gate_w,23 cm_lin_b,
  // 24 cm_lin_w,25 ku_b1,26 ku_b2,27 ku_w1,28 ku_w2,29 ml_b,30 ml_ln_b,31 ml_ln_g,
  // 32 ml_w,33 mu_b1,34 mu_b2,35 mu_w1,36 mu_w2,37 skip_b,38 skip_w,39 we,40 we_b,
  // 41 wk,42 wk_b,43 wq,44 wq_b,45 wv,46 wv_b, 47 rbf_b1,48 rbf_b2,49 rbf_w1,50 rbf_w2

  char* wsb = (char*)d_ws; size_t off = 0;
  auto alloc = [&](size_t bytes) -> void* {
    void* p = wsb + off; off = (off + bytes + 255) & ~(size_t)255; return p;
  };
  auto AF = [&](size_t n) { return (float*)alloc(n * 4); };
  auto AB = [&](size_t n) { return (bf16*)alloc(n * 2); };

  float* x_c = AF((size_t)N * H); float* x_m = AF((size_t)N * H);
  float* x_cart = AF((size_t)N * H); float* x_mat = AF((size_t)N * H);
  float* x_out = AF((size_t)N * H); float* x_r = AF((size_t)N * H);
  float* agg = AF((size_t)N * H);  float* outn = AF((size_t)N * H);
  float* gate2 = AF((size_t)N * H); float* fusedb = AF((size_t)N * H);
  float* qbuf = AF((size_t)N * HC);
  float* e_f = AF((size_t)E * H);
  float* gate = AF((size_t)E * H); float* msgc = AF((size_t)E * H);
  float* keyb = AF((size_t)E * 4 * H);   // also reused as ml-GEMM tmp
  float* msg  = AF((size_t)E * 4 * H);
  float* meanb = AF(H); float* rstdb = AF(H);
  float* mean2 = AF(H); float* rstd2 = AF(H);
  float* psums = AF((size_t)NG * H); float* pcnt = AF(NG);

  bf16* x_c_bf = AB((size_t)N * H); bf16* x_m_bf = AB((size_t)N * H);
  bf16* xx_bf = AB((size_t)N * 2 * H);
  bf16* e_bf = AB((size_t)E * H);
  bf16* rbf_bf = AB((size_t)E * H);
  bf16* hidE_bf = AB((size_t)E * H);
  bf16* hid4_bf = AB((size_t)E * 4 * H);
  bf16* k_bf = AB((size_t)N * HC); bf16* v_bf = AB((size_t)N * HC);
  bf16* ee_bf = AB((size_t)E * HC);
  bf16* msg_bf = AB((size_t)E * 4 * H);
  bf16* rbfw1T = AB((size_t)H * H); bf16* rbfw2T = AB((size_t)H * H);

  auto T = [&](const float* W, bf16* WT, int K, int Nc) {
    long long tot = (long long)K * Nc;
    k_transpose<<<dim3((unsigned)((tot + 255) / 256)), 256, 0, stream>>>(W, WT, K, Nc);
  };
  auto GEMM = [&](const bf16* A, int lda, const bf16* WT, const float* bias,
                  float* C, bf16* Cbf, int ldc, int M, int Nc, int K, int act) {
    dim3 g((unsigned)(M / 16), (unsigned)((Nc + 127) / 128));
    k_gemm<<<g, 128, 0, stream>>>(A, lda, WT, bias, C, Cbf, ldc, Nc, K, act);
  };
  auto GEMM3 = [&](const bf16* X0, int ld0, const bf16* X1, int ld1,
                   const bf16* X2, int ld2, const int* i0, const int* i1, int hshift,
                   const bf16* WT, const float* bias, float* C, bf16* Cbf, int ldc,
                   int M, int Nc, int act) {
    dim3 g((unsigned)(M / 16), (unsigned)((Nc + 127) / 128));
    k_gemm_cat3<<<g, 128, 0, stream>>>(X0, ld0, X1, ld1, X2, ld2, i0, i1, hshift,
                                       WT, bias, C, Cbf, ldc, Nc, act);
  };
  auto ZERO = [&](float* p, long long n) {
    k_zero<<<dim3((unsigned)((n + 255) / 256)), 256, 0, stream>>>(p, n);
  };

  // ---- per-layer transposed bf16 weights -----------------------------------
  struct LayerW { bf16 *cgw1,*cgw2,*caw1,*caw2,*wq,*wk,*wv,*we,*kuw1,*kuw2,*muw1,*muw2,*mlw,*skip,*cmg,*cml; };
  LayerW lw[5];
  for (int l = 0; l < L; ++l) {
    lw[l].cgw1 = AB(384 * 128); T(F(19) + (size_t)l * 384 * 128, lw[l].cgw1, 384, 128);
    lw[l].cgw2 = AB(128 * 128); T(F(20) + (size_t)l * 128 * 128, lw[l].cgw2, 128, 128);
    lw[l].caw1 = AB(384 * 128); T(F(15) + (size_t)l * 384 * 128, lw[l].caw1, 384, 128);
    lw[l].caw2 = AB(128 * 128); T(F(16) + (size_t)l * 128 * 128, lw[l].caw2, 128, 128);
    lw[l].wq = AB(128 * 512); T(F(43) + (size_t)l * 128 * 512, lw[l].wq, 128, 512);
    lw[l].wk = AB(128 * 512); T(F(41) + (size_t)l * 128 * 512, lw[l].wk, 128, 512);
    lw[l].wv = AB(128 * 512); T(F(45) + (size_t)l * 128 * 512, lw[l].wv, 128, 512);
    lw[l].we = AB(128 * 512); T(F(39) + (size_t)l * 128 * 512, lw[l].we, 128, 512);
    lw[l].kuw1 = AB(384 * 128); T(F(27) + (size_t)l * 384 * 128, lw[l].kuw1, 384, 128);
    lw[l].kuw2 = AB(128 * 128); T(F(28) + (size_t)l * 128 * 128, lw[l].kuw2, 128, 128);
    lw[l].muw1 = AB(384 * 128); T(F(35) + (size_t)l * 384 * 128, lw[l].muw1, 384, 128);
    lw[l].muw2 = AB(128 * 128); T(F(36) + (size_t)l * 128 * 128, lw[l].muw2, 128, 128);
    lw[l].mlw  = AB(128 * 128); T(F(32) + (size_t)l * 128 * 128, lw[l].mlw, 128, 128);
    lw[l].skip = AB(128 * 128); T(F(38) + (size_t)l * 128 * 128, lw[l].skip, 128, 128);
    lw[l].cmg  = AB(256 * 128); T(F(22) + (size_t)l * 256 * 128, lw[l].cmg, 256, 128);
    lw[l].cml  = AB(256 * 128); T(F(24) + (size_t)l * 256 * 128, lw[l].cml, 256, 128);
  }
  T(F(49), rbfw1T, 128, 128);
  T(F(50), rbfw2T, 128, 128);

  // ---- setup: embedding + RBF edge encoder --------------------------------
  k_embed<<<dim3((unsigned)(N * H / 256)), 256, 0, stream>>>(xtype, F(4), x_c, x_m, x_c_bf, x_m_bf, N);
  k_rbf<<<dim3((unsigned)((long long)E * H / 256)), 256, 0, stream>>>(eattr, rbf_bf, E);
  GEMM(rbf_bf, H, rbfw1T, F(47), nullptr, hidE_bf, H, E, H, H, 3 /*softplus*/);
  GEMM(hidE_bf, H, rbfw2T, F(48), e_f, e_bf, H, E, H, H, 0);

  // ---- layers --------------------------------------------------------------
  for (int l = 0; l < L; ++l) {
    const LayerW& w = lw[l];
    const float* cg_b1 = F(17) + l * H; const float* cg_b2 = F(18) + l * H;
    const float* ca_b1 = F(13) + l * H; const float* ca_b2 = F(14) + l * H;
    const float* wq_b = F(44) + l * HC; const float* wk_b = F(42) + l * HC;
    const float* wv_b = F(46) + l * HC; const float* we_b = F(40) + l * HC;
    const float* ku_b1 = F(25) + l * H; const float* ku_b2 = F(26) + l * H;
    const float* mu_b1 = F(33) + l * H; const float* mu_b2 = F(34) + l * H;
    const float* ml_b = F(29) + l * H;
    const float* ml_ln_g = F(31) + l * H; const float* ml_ln_b = F(30) + l * H;
    const float* a_ln_g = F(10) + l * H;  const float* a_ln_b = F(9) + l * H;
    const float* skip_b = F(37) + l * H;
    const float* beta_w = F(12) + l * 384; const float* beta_b = F(11) + l;
    const float* cm_gate_b = F(21) + l * H; const float* cm_lin_b = F(23) + l * H;

    // CartNet gated message passing (uses pre-update x_c, e)
    GEMM3(x_c_bf, H, x_c_bf, H, e_bf, H, dstI, srcI, 0, w.cgw1, cg_b1, nullptr, hidE_bf, H, E, H, 1);
    GEMM(hidE_bf, H, w.cgw2, cg_b2, gate, nullptr, H, E, H, H, 0);
    GEMM3(x_c_bf, H, x_c_bf, H, e_bf, H, dstI, srcI, 0, w.caw1, ca_b1, nullptr, hidE_bf, H, E, H, 1);
    GEMM(hidE_bf, H, w.caw2, ca_b2, msgc, nullptr, H, E, H, H, 0);
    k_colstats<<<128, 256, 0, stream>>>(gate, E, meanb, rstdb);
    ZERO(agg, (long long)N * H);
    k_cart_edge<<<dim3((unsigned)((long long)E * H / 256)), 256, 0, stream>>>(
        gate, msgc, dstI, meanb, rstdb, e_f, e_bf, agg, E);
    k_colstats<<<128, 256, 0, stream>>>(agg, N, mean2, rstd2);
    k_bn_silu<<<dim3((unsigned)(N * H / 256)), 256, 0, stream>>>(x_c, agg, mean2, rstd2, x_cart, N);

    // Matformer edge-conditioned attention (uses x_m, e_new)
    GEMM(x_m_bf, H, w.wq, wq_b, qbuf, nullptr, HC, N, HC, H, 0);
    GEMM(x_m_bf, H, w.wk, wk_b, nullptr, k_bf, HC, N, HC, H, 0);
    GEMM(x_m_bf, H, w.wv, wv_b, nullptr, v_bf, HC, N, HC, H, 0);
    GEMM(e_bf, H, w.we, we_b, nullptr, ee_bf, HC, E, HC, H, 0);
    GEMM3(k_bf, HC, k_bf, HC, ee_bf, HC, dstI, srcI, 2, w.kuw1, ku_b1, nullptr, hid4_bf, H, E * 4, H, 1);
    GEMM(hid4_bf, H, w.kuw2, ku_b2, keyb, nullptr, H, E * 4, H, H, 0);
    GEMM3(v_bf, HC, v_bf, HC, ee_bf, HC, dstI, srcI, 2, w.muw1, mu_b1, nullptr, hid4_bf, H, E * 4, H, 1);
    GEMM(hid4_bf, H, w.muw2, mu_b2, msg, nullptr, H, E * 4, H, H, 0);
    k_alpha<<<dim3((unsigned)(E * 4)), 128, 0, stream>>>(qbuf, msg, msg_bf, keyb, dstI, a_ln_g, a_ln_b);
    GEMM(msg_bf, H, w.mlw, ml_b, keyb /*tmp*/, nullptr, H, E * 4, H, H, 0);
    ZERO(outn, (long long)N * H);
    k_mlout<<<dim3((unsigned)(E * 4)), 128, 0, stream>>>(keyb, dstI, ml_ln_g, ml_ln_b, outn);
    GEMM(x_m_bf, H, w.skip, skip_b, x_r, nullptr, H, N, H, H, 0);
    k_beta<<<dim3((unsigned)N), 128, 0, stream>>>(outn, x_r, beta_w, beta_b, x_mat);

    // CrossMix fusion + carry update
    k_concat2<<<dim3((unsigned)(N * H / 256)), 256, 0, stream>>>(x_cart, x_mat, xx_bf, N);
    GEMM(xx_bf, 2 * H, w.cmg, cm_gate_b, gate2, nullptr, H, N, H, 2 * H, 0);
    GEMM(xx_bf, 2 * H, w.cml, cm_lin_b, fusedb, nullptr, H, N, H, 2 * H, 0);
    k_mix<<<dim3((unsigned)(N * H / 256)), 256, 0, stream>>>(
        gate2, fusedb, x_cart, x_mat, x_out, x_c, x_m, x_c_bf, x_m_bf, N);
  }

  // ---- readout -------------------------------------------------------------
  ZERO(psums, (long long)NG * H);
  ZERO(pcnt, NG);
  k_pool<<<dim3((unsigned)(N * H / 256)), 256, 0, stream>>>(x_out, batch, psums, pcnt, N);
  k_fc<<<dim3((unsigned)NG), 128, 0, stream>>>(psums, pcnt, F(7), F(5), F(8), F(6), (float*)d_out);
}
